// MultiheadAttention_36919538876914
// MI455X (gfx1250) — matrix operand order repaired and checked by reading, not compiled, not run on hardware
//
#include <hip/hip_runtime.h>
#include <hip/hip_bf16.h>
#include <cstdint>
#include <cstddef>

#define BSZ 4
#define NH 12
#define SEQ 1024
#define HDIM 64
#define DM 768

typedef __attribute__((ext_vector_type(16))) _Float16 v16h;
typedef __attribute__((ext_vector_type(8)))  _Float16 v8h;
typedef __attribute__((ext_vector_type(8)))  float    v8f;

static __device__ __forceinline__ v8f wmma_f16(v16h a, v16h b, v8f c) {
  // D = A(16x32 f16) * B(32x16 f16) + C(16x16 f32)
  return __builtin_amdgcn_wmma_f32_16x16x32_f16(false, a, false, b, (short)0, c,
                                                false, false);
}

// ---------------------------------------------------------------------------
// Kernel 0: biasT[h][q][k] = spatial[q][k][h] + edge[h][q][k]
// ---------------------------------------------------------------------------
__global__ __launch_bounds__(256) void bias_combine_kernel(
    const float* __restrict__ spatial, const float* __restrict__ edge,
    float* __restrict__ biasT) {
  const int idx = blockIdx.x * 256 + threadIdx.x;  // idx = q*SEQ + k
  const float* sp = spatial + (size_t)idx * NH;
  float s[NH];
#pragma unroll
  for (int h = 0; h < NH; ++h) s[h] = sp[h];
#pragma unroll
  for (int h = 0; h < NH; ++h) {
    const size_t o = (size_t)h * SEQ * SEQ + (size_t)idx;
    biasT[o] = s[h] + edge[o];
  }
}

// ---------------------------------------------------------------------------
// Kernels 1-3: y = (x @ W^T + b) * scale -> f16. One wave per 16x64 tile:
// one A-fragment feeds 4 WMMAs (4x less A traffic, denser XDL issue).
// transposeV=0: out[b][h][l][c]; transposeV=1: out[b][h][c][l]
// ---------------------------------------------------------------------------
__global__ __launch_bounds__(128) void proj_qkv_kernel(
    const float* __restrict__ x, const float* __restrict__ W,
    const float* __restrict__ bias, _Float16* __restrict__ out,
    const int transposeV, const float scale) {
  const int wid  = blockIdx.x * 4 + (threadIdx.x >> 5);  // 0..3071
  const int nt   = wid % (DM / 64);                      // 64-wide col group
  const int mt   = wid / (DM / 64);                      // 16-row group
  const int lane = threadIdx.x & 31;
  const int half = lane >> 4;
  const int mn   = lane & 15;

  const float* arow = x + (size_t)(mt * 16 + mn) * DM;
  const float* brow = W + (size_t)(nt * 64 + mn) * DM;   // + ct*16*DM per ct

  v8f acc[4] = {{}, {}, {}, {}};
  for (int kk = 0; kk < DM; kk += 32) {
    v16h a;
    const float* a1 = arow + kk + 16 * half;       // A elems 0..7 : K=16h..16h+7
    const float* a2 = a1 + 8;                      // A elems 8..15: K=16h+8..
#pragma unroll
    for (int i = 0; i < 8; ++i) { a[i] = (_Float16)a1[i]; a[8 + i] = (_Float16)a2[i]; }
#pragma unroll
    for (int ct = 0; ct < 4; ++ct) {
      const float* bp = brow + (size_t)ct * 16 * DM + kk + 16 * half;
      v16h bfrag;
#pragma unroll
      for (int i = 0; i < 16; ++i) bfrag[i] = (_Float16)bp[i];
      acc[ct] = wmma_f16(a, bfrag, acc[ct]);
    }
  }

#pragma unroll
  for (int ct = 0; ct < 4; ++ct) {
    const int o = nt * 64 + ct * 16 + mn;          // output feature (col)
    const int h = o / HDIM, c = o % HDIM;
    const float bia = bias[o];
#pragma unroll
    for (int r = 0; r < 8; ++r) {                  // C/D row = r + 8*half
      const int m  = mt * 16 + r + 8 * half;       // global row = b*SEQ + l
      const int bb = m >> 10, l = m & (SEQ - 1);
      const float v = (acc[ct][r] + bia) * scale;
      if (transposeV)
        out[((size_t)(bb * NH + h) * HDIM + c) * SEQ + l] = (_Float16)v;
      else
        out[((size_t)(bb * NH + h) * SEQ + l) * HDIM + c] = (_Float16)v;
    }
  }
}

// ---------------------------------------------------------------------------
// Kernel 4: fused attention. 4 waves per workgroup share one 16-row q-block.
//   Pass1: waves stripe k-tiles; S = (Q/8)K^T + bias -> 64KB LDS.
//   Stats: each wave reduces 4 rows (float4 + shfl_xor wave32 tree).
//   Pass2: waves split k-windows; P -> NT f32 attention store + f16 A-frag;
//          partial O via 4 WMMA/window, reduced across waves with ds_add_f32.
// LDS ~68KB -> 4 workgroups (16 waves) per 320KB WGP.
// ---------------------------------------------------------------------------
__global__ __launch_bounds__(128) void attn_kernel(
    const _Float16* __restrict__ qf, const _Float16* __restrict__ kf,
    const _Float16* __restrict__ vT, const float* __restrict__ biasT,
    float* __restrict__ attn, _Float16* __restrict__ ctx) {
  __shared__ float srow[16 * SEQ];   // 64 KB score row-block
  __shared__ float stats[32];        // [0..15]=rowmax, [16..31]=1/rowsum
  __shared__ float obuf[16 * HDIM];  // 4 KB cross-wave O reduction

  const int wg = blockIdx.x;
  const int qb = wg % (SEQ / 16);
  const int bh = wg / (SEQ / 16);
  const int h  = bh % NH;
  const int b  = bh / NH;
  const int tid  = threadIdx.x;
  const int wave = tid >> 5;
  const int lane = tid & 31;
  const int half = lane >> 4;
  const int mn   = lane & 15;

#pragma unroll
  for (int i = 0; i < 8; ++i) obuf[tid + i * 128] = 0.f;

  // Q fragments (A: lane row = mn, c = 0..63); identical in all 4 waves
  const _Float16* qrow = qf + ((size_t)bh * SEQ + qb * 16 + mn) * HDIM;
  v16h a0, a1;
  {
    const v8h r0 = *(const v8h*)(qrow +  0 + 16 * half);
    const v8h r1 = *(const v8h*)(qrow +  8 + 16 * half);
    const v8h r2 = *(const v8h*)(qrow + 32 + 16 * half);
    const v8h r3 = *(const v8h*)(qrow + 40 + 16 * half);
#pragma unroll
    for (int i = 0; i < 8; ++i) {
      a0[i] = r0[i]; a0[8 + i] = r1[i];
      a1[i] = r2[i]; a1[8 + i] = r3[i];
    }
  }

  // ---- Pass 1: scores + bias into LDS (waves stripe k-tiles) ----
  const float* bias_base =
      biasT + (size_t)h * SEQ * SEQ + (size_t)(qb * 16) * SEQ;
  for (int kt = wave; kt < SEQ / 16; kt += 4) {
    const _Float16* krow =
        kf + ((size_t)bh * SEQ + kt * 16 + mn) * HDIM + 16 * half;
    const v16h b0 = *(const v16h*)(krow);        // B = K^T, c = 0..31
    const v16h b1 = *(const v16h*)(krow + 32);   //           c = 32..63
    v8f s = {};
    s = wmma_f16(a0, b0, s);
    s = wmma_f16(a1, b1, s);
    const int k = kt * 16 + mn;
#pragma unroll
    for (int r = 0; r < 8; ++r) {
      const int row = r + 8 * half;
      srow[row * SEQ + k] = s[r] + bias_base[(size_t)row * SEQ + k];
    }
  }
  __syncthreads();

  // ---- row stats: wave w reduces rows 4w..4w+3 ----
#pragma unroll
  for (int rr = 0; rr < 4; ++rr) {
    const int row = wave * 4 + rr;
    const float* rp = srow + row * SEQ;
    float mx = -3.0e38f;
#pragma unroll
    for (int j = 0; j < 8; ++j) {
      const float4 t = *(const float4*)(rp + lane * 4 + j * 128);
      mx = fmaxf(mx, fmaxf(fmaxf(t.x, t.y), fmaxf(t.z, t.w)));
    }
    mx = fmaxf(mx, __shfl_xor(mx, 16, 32));
    mx = fmaxf(mx, __shfl_xor(mx, 8, 32));
    mx = fmaxf(mx, __shfl_xor(mx, 4, 32));
    mx = fmaxf(mx, __shfl_xor(mx, 2, 32));
    mx = fmaxf(mx, __shfl_xor(mx, 1, 32));
    float sum = 0.f;
#pragma unroll
    for (int j = 0; j < 8; ++j) {
      const float4 t = *(const float4*)(rp + lane * 4 + j * 128);
      sum += __expf(t.x - mx) + __expf(t.y - mx) +
             __expf(t.z - mx) + __expf(t.w - mx);
    }
    sum += __shfl_xor(sum, 16, 32);
    sum += __shfl_xor(sum, 8, 32);
    sum += __shfl_xor(sum, 4, 32);
    sum += __shfl_xor(sum, 2, 32);
    sum += __shfl_xor(sum, 1, 32);
    if (lane == 0) { stats[row] = mx; stats[16 + row] = 1.0f / sum; }
  }
  __syncthreads();
  const float m_   = stats[mn];
  const float inv_ = stats[16 + mn];

  // ---- Pass 2: softmax out + partial O = P @ V (waves split k-range) ----
  float* arow_out = attn + ((size_t)bh * SEQ + qb * 16 + mn) * SEQ;
  v8f o0 = {}, o1 = {}, o2 = {}, o3 = {};
  for (int kw = wave * (SEQ / 4); kw < (wave + 1) * (SEQ / 4); kw += 32) {
    v16h ap;                                     // A-frag of P: row mn
    const int k1 = kw + 16 * half;
    const int k2 = k1 + 8;
#pragma unroll
    for (int i = 0; i < 8; ++i) {
      const float p1 = __expf(srow[mn * SEQ + k1 + i] - m_) * inv_;
      const float p2 = __expf(srow[mn * SEQ + k2 + i] - m_) * inv_;
      ap[i]     = (_Float16)p1;
      ap[8 + i] = (_Float16)p2;
      __builtin_nontemporal_store(p1, arow_out + k1 + i);  // 201MB stream: NT
      __builtin_nontemporal_store(p2, arow_out + k2 + i);
    }
    const _Float16* vbase = vT + (size_t)bh * HDIM * SEQ + kw + 16 * half;
    const v16h bv0 = *(const v16h*)(vbase + (size_t)( 0 + mn) * SEQ);
    const v16h bv1 = *(const v16h*)(vbase + (size_t)(16 + mn) * SEQ);
    const v16h bv2 = *(const v16h*)(vbase + (size_t)(32 + mn) * SEQ);
    const v16h bv3 = *(const v16h*)(vbase + (size_t)(48 + mn) * SEQ);
    o0 = wmma_f16(ap, bv0, o0);
    o1 = wmma_f16(ap, bv1, o1);
    o2 = wmma_f16(ap, bv2, o2);
    o3 = wmma_f16(ap, bv3, o3);
  }

  // ---- cross-wave O reduction (ds_add_f32) ----
#pragma unroll
  for (int r = 0; r < 8; ++r) {
    const int row = r + 8 * half;
    atomicAdd(&obuf[row * HDIM +  0 + mn], o0[r]);
    atomicAdd(&obuf[row * HDIM + 16 + mn], o1[r]);
    atomicAdd(&obuf[row * HDIM + 32 + mn], o2[r]);
    atomicAdd(&obuf[row * HDIM + 48 + mn], o3[r]);
  }
  __syncthreads();

  // ---- ctx[b][l][h*64+c] (f16), 1024 values over 128 threads ----
#pragma unroll
  for (int i = 0; i < 8; ++i) {
    const int idx = tid + i * 128;
    const int row = idx >> 6, c = idx & (HDIM - 1);
    ctx[((size_t)b * SEQ + qb * 16 + row) * DM + h * HDIM + c] =
        (_Float16)obuf[idx];
  }
}

// ---------------------------------------------------------------------------
// Kernel 5: out = ctx(f16) @ Wo^T + bo, fp32. One wave per 16x64 tile.
// ---------------------------------------------------------------------------
__global__ __launch_bounds__(128) void out_proj_kernel(
    const _Float16* __restrict__ ctx, const float* __restrict__ W,
    const float* __restrict__ bias, float* __restrict__ out) {
  const int wid  = blockIdx.x * 4 + (threadIdx.x >> 5);
  const int nt   = wid % (DM / 64);
  const int mt   = wid / (DM / 64);
  const int lane = threadIdx.x & 31;
  const int half = lane >> 4;
  const int mn   = lane & 15;

  const _Float16* arow = ctx + (size_t)(mt * 16 + mn) * DM;
  const float*    brow = W + (size_t)(nt * 64 + mn) * DM;

  v8f acc[4] = {{}, {}, {}, {}};
  for (int kk = 0; kk < DM; kk += 32) {
    v16h a;
    const v8h r1 = *(const v8h*)(arow + kk + 16 * half);
    const v8h r2 = *(const v8h*)(arow + kk + 8 + 16 * half);
#pragma unroll
    for (int i = 0; i < 8; ++i) { a[i] = r1[i]; a[8 + i] = r2[i]; }
#pragma unroll
    for (int ct = 0; ct < 4; ++ct) {
      const float* bp = brow + (size_t)ct * 16 * DM + kk + 16 * half;
      v16h bfrag;
#pragma unroll
      for (int i = 0; i < 16; ++i) bfrag[i] = (_Float16)bp[i];
      acc[ct] = wmma_f16(a, bfrag, acc[ct]);
    }
  }

#pragma unroll
  for (int ct = 0; ct < 4; ++ct) {
    const int o = nt * 64 + ct * 16 + mn;
    const float bo = bias[o];
#pragma unroll
    for (int r = 0; r < 8; ++r) {
      const int m = mt * 16 + r + 8 * half;
      out[(size_t)m * DM + o] = acc[ct][r] + bo;
    }
  }
}

// ---------------------------------------------------------------------------
extern "C" void kernel_launch(void* const* d_in, const int* in_sizes, int n_in,
                              void* d_out, int out_size, void* d_ws,
                              size_t ws_size, hipStream_t stream) {
  (void)in_sizes; (void)n_in; (void)out_size; (void)ws_size;
  const float* x       = (const float*)d_in[0];
  const float* spatial = (const float*)d_in[1];
  const float* edge    = (const float*)d_in[2];
  const float* Wq = (const float*)d_in[3];
  const float* bq = (const float*)d_in[4];
  const float* Wk = (const float*)d_in[5];
  const float* bk = (const float*)d_in[6];
  const float* Wv = (const float*)d_in[7];
  const float* bv = (const float*)d_in[8];
  const float* Wo = (const float*)d_in[9];
  const float* bo = (const float*)d_in[10];

  float* out  = (float*)d_out;                       // [B,L,D] fp32
  float* attn = out + (size_t)BSZ * SEQ * DM;        // [B,H,L,L] fp32

  // scratch: biasT (50.3 MB) | q | k | vT | ctx (f16, 6.3 MB each)
  float*    biasT = (float*)d_ws;
  _Float16* qf    = (_Float16*)(biasT + (size_t)NH * SEQ * SEQ);
  const size_t per = (size_t)BSZ * SEQ * DM;
  _Float16* kfp = qf + per;
  _Float16* vT  = kfp + per;
  _Float16* ctx = vT + per;

  const int gemm_blocks = (BSZ * SEQ / 16) * (DM / 64) / 4;  // 768

  bias_combine_kernel<<<SEQ * SEQ / 256, 256, 0, stream>>>(spatial, edge, biasT);
  proj_qkv_kernel<<<gemm_blocks, 128, 0, stream>>>(x, Wq, bq, qf,  0, 0.125f);
  proj_qkv_kernel<<<gemm_blocks, 128, 0, stream>>>(x, Wk, bk, kfp, 0, 1.0f);
  proj_qkv_kernel<<<gemm_blocks, 128, 0, stream>>>(x, Wv, bv, vT,  1, 1.0f);
  attn_kernel<<<BSZ * NH * (SEQ / 16), 128, 0, stream>>>(qf, kfp, vT, biasT,
                                                         attn, ctx);
  out_proj_kernel<<<gemm_blocks, 128, 0, stream>>>(ctx, Wo, bo, out);
}
